// RuchbahDynamicH2OAttention_4131758538900
// MI455X (gfx1250) — matrix operand order repaired and checked by reading, not compiled, not run on hardware
//
#include <hip/hip_runtime.h>
#include <hip/hip_bf16.h>

typedef __attribute__((ext_vector_type(16))) __bf16 v16bf;
typedef __attribute__((ext_vector_type(8)))  float  v8f;
typedef __attribute__((ext_vector_type(4)))  unsigned int u32x4;
typedef __attribute__((ext_vector_type(8)))  int i32x8;
typedef __attribute__((ext_vector_type(4)))  int i32x4;

struct Frag32 { uint4 a, b; };

// Build a 16-element bf16 fragment from two 16-byte chunks.
__device__ __forceinline__ v16bf frag2(const unsigned short* p0, const unsigned short* p1) {
    Frag32 f;
    f.a = *(const uint4*)p0;
    f.b = *(const uint4*)p1;
    return __builtin_bit_cast(v16bf, f);
}

__device__ __forceinline__ unsigned short f2bf(float x) {
    __bf16 h = (__bf16)x;                       // hw bf16 convert path
    return __builtin_bit_cast(unsigned short, h);
}

__device__ __forceinline__ unsigned pack2(float a, float b) {
    return ((unsigned)f2bf(b) << 16) | (unsigned)f2bf(a);
}

// Convert 4 f32 -> 4 bf16, one 8-byte LDS store.
__device__ __forceinline__ void cvt_store4(unsigned short* d, float4 v) {
    uint2 u;
    u.x = pack2(v.x, v.y);
    u.y = pack2(v.z, v.w);
    *(uint2*)d = u;
}

__device__ __forceinline__ v8f wmma_bf16(v16bf a, v16bf b, v8f c) {
    return __builtin_amdgcn_wmma_f32_16x16x32_bf16(false, a, false, b, (short)0, c, false, false);
}

__device__ __forceinline__ v8f zero8() {
    v8f z = {0.f, 0.f, 0.f, 0.f, 0.f, 0.f, 0.f, 0.f};
    return z;
}

// ---------------------------------------------------------------------------
// TDM: 2D bf16 tile load (global -> LDS) via Tensor Data Mover (D# per ISA
// 8.3-8.6: count=1, type=2 "image", data_size code 1 = 2 bytes).
// This toolchain exposes the 6-arg builtin (5 descriptor groups + cpol).
// ---------------------------------------------------------------------------
__device__ __forceinline__ void tdm_load_tile_bf16(unsigned lds_off, const void* gptr,
                                                   unsigned tile_d0, unsigned tile_d1,
                                                   unsigned tensor_d0, unsigned stride_elems) {
    unsigned long long ga = (unsigned long long)(uintptr_t)gptr;
    u32x4 g0;
    g0.x = 1u;                                              // count=1 (valid user D#)
    g0.y = lds_off;                                         // lds_addr (bytes)
    g0.z = (unsigned)ga;                                    // global_addr[31:0]
    g0.w = (unsigned)((ga >> 32) & 0x01FFFFFFu) | (2u << 30);  // addr[56:32] | type=2
    i32x8 g1;
    g1[0] = (int)(1u << 16);                                // data_size=1 -> 2B elements
    g1[1] = (int)((tensor_d0 & 0xFFFFu) << 16);             // tensor_dim0[15:0]
    g1[2] = (int)(((tensor_d0 >> 16) & 0xFFFFu) | (0xFFFFu << 16)); // dim0 hi | dim1 lo
    g1[3] = (int)((tile_d0 & 0xFFFFu) << 16);               // dim1 hi=0 | tile_dim0
    g1[4] = (int)(tile_d1 & 0xFFFFu);                       // tile_dim1 (tile_dim2=0)
    g1[5] = (int)stride_elems;                              // tensor_dim0_stride[31:0]
    g1[6] = 0;
    g1[7] = 0;
    i32x4 gz4 = {0, 0, 0, 0};
    i32x8 gz8 = {0, 0, 0, 0, 0, 0, 0, 0};
    __builtin_amdgcn_tensor_load_to_lds(g0, g1, gz4, gz4, gz8, 0);
}

// ---------------------------------------------------------------------------
// GEMM: C[M,N] = A[M,K] * B[N,K]^T  (x @ W.T), bf16 compute, f32 accum.
// MODE 0: C float [M,N].  MODE 1: C bf16 remapped to [B*nh][S][hd] head layout.
// 256 threads = 8 waves; 128x128 tile; K-step 32; double-buffered LDS.
// ---------------------------------------------------------------------------
#define AST 40  // LDS row stride in bf16 elems (80B, 16B-aligned, bank-skewed)

template<int MODE>
__global__ __launch_bounds__(256)
void gemm_xwt_kernel(const float* __restrict__ A, const float* __restrict__ B,
                     void* __restrict__ Cout, int M, int N, int K) {
    __shared__ unsigned short Ash[2][128 * AST];
    __shared__ unsigned short Bsh[2][128 * AST];

    const int tid  = threadIdx.x;
    const int lane = tid & 31, wave = tid >> 5;
    const int l16  = lane & 15, half = lane >> 4;
    const int wm   = wave & 3,  wn   = wave >> 2;
    const int m0   = blockIdx.y * 128;
    const int n0   = blockIdx.x * 128;

    v8f acc[2][4];
    for (int i = 0; i < 2; ++i)
        for (int j = 0; j < 4; ++j) acc[i][j] = zero8();

    auto stage = [&](int kt, int buf) {
        for (int i = 0; i < 4; ++i) {
            int f = tid + i * 256;            // 0..1023 float4 slots
            int r = f >> 3, c = (f & 7) << 2; // row, 4-col group
            float4 av = *(const float4*)(A + (size_t)(m0 + r) * K + kt + c);
            cvt_store4(&Ash[buf][r * AST + c], av);
            float4 bv = *(const float4*)(B + (size_t)(n0 + r) * K + kt + c);
            cvt_store4(&Bsh[buf][r * AST + c], bv);
        }
    };

    stage(0, 0);
    __syncthreads();

    const int nk = K >> 5;
    for (int kt = 0; kt < nk; ++kt) {
        const int buf = kt & 1;
        if (kt + 1 < nk) stage((kt + 1) << 5, buf ^ 1);   // prefetch next tile

        v16bf af[2], bfr[4];
        for (int mt = 0; mt < 2; ++mt) {
            const unsigned short* p = &Ash[buf][(wm * 32 + mt * 16 + l16) * AST + 8 * half];
            af[mt] = frag2(p, p + 16);  // A layout: k = 16*(j/8) + 8*half + j%8
        }
        for (int nt = 0; nt < 4; ++nt) {
            const unsigned short* p = &Bsh[buf][(wn * 64 + nt * 16 + l16) * AST + 8 * half];
            bfr[nt] = frag2(p, p + 16);  // B layout: k = 16*(j/8) + 8*half + j%8
        }
        for (int mt = 0; mt < 2; ++mt)
            for (int nt = 0; nt < 4; ++nt)
                acc[mt][nt] = wmma_bf16(af[mt], bfr[nt], acc[mt][nt]);
        __syncthreads();
    }

    // Epilogue. C layout: VGPR r holds row (r + 8*half), lane l16 holds col.
    for (int mt = 0; mt < 2; ++mt)
        for (int nt = 0; nt < 4; ++nt)
            for (int r = 0; r < 8; ++r) {
                int m = m0 + wm * 32 + mt * 16 + r + 8 * half;
                int n = n0 + wn * 64 + nt * 16 + l16;
                float v = acc[mt][nt][r];
                if (MODE == 0) {
                    ((float*)Cout)[(size_t)m * N + n] = v;
                } else {
                    int b = m >> 11, s = m & 2047;      // S = 2048
                    int h = n >> 7,  d = n & 127;       // hd = 128
                    size_t idx = ((((size_t)(b * 16 + h) << 11) | s) << 7) | d;
                    ((unsigned short*)Cout)[idx] = f2bf(v);
                }
            }
}

// ---------------------------------------------------------------------------
// Flash attention: one block per (b*nh, 128-query tile). 8 waves x 16 q-rows.
// Q/K/V bf16 [B*nh][S][128]; output f32 [B,S,H]; softmax in exp2 domain.
// K tile staged by the Tensor Data Mover; V staged+transposed by VALU.
// ---------------------------------------------------------------------------
#define HS  128
#define KST 136  // padded LDS stride for Vt/Psh (272B, 16B-aligned)

__global__ __launch_bounds__(256)
void attn_kernel(const unsigned short* __restrict__ Qb,
                 const unsigned short* __restrict__ Kb,
                 const unsigned short* __restrict__ Vb,
                 float* __restrict__ Ob, int S, int nh) {
    __shared__ unsigned short Ksh[128 * HS];        // [key][hd]   (TDM target)
    __shared__ unsigned short Vt [128 * KST];       // [hd][key]   (transposed)
    __shared__ unsigned short Psh[8 * 16 * KST];    // per-wave P scratch

    const int tid  = threadIdx.x;
    const int lane = tid & 31, wave = tid >> 5;
    const int l16  = lane & 15, half = lane >> 4;
    const int nqt  = S / 128;
    const int bh   = blockIdx.x / nqt;
    const int qt   = blockIdx.x % nqt;
    const size_t hbase = (size_t)bh * S * HS;
    const int qrow = qt * 128 + wave * 16;

    // Q fragments stay in registers for the whole kernel (A-matrix layout).
    v16bf qf[4];
    for (int ks = 0; ks < 4; ++ks) {
        const unsigned short* p = Qb + hbase + (size_t)(qrow + l16) * HS + ks * 32 + 8 * half;
        qf[ks] = frag2(p, p + 16);
    }

    const float SL = 0.08838834764831845f * 1.4426950408889634f;  // (1/sqrt(128))*log2(e)

    float mrow[8], lrow[8];
    v8f oacc[8];
    for (int r = 0; r < 8; ++r) { mrow[r] = -3.0e38f; lrow[r] = 0.f; }
    for (int ot = 0; ot < 8; ++ot) oacc[ot] = zero8();

    const int nkb = S / 128;
    for (int kb = 0; kb < nkb; ++kb) {
        __syncthreads();
        // K tile via TDM (wave 0 issues the DMA descriptor and waits TENSORcnt).
        if (wave == 0) {
            tdm_load_tile_bf16((unsigned)(size_t)(void*)Ksh,
                               Kb + hbase + (size_t)(kb * 128) * HS,
                               HS, 128, HS, HS);
            __builtin_amdgcn_s_wait_tensorcnt(0);
        }
        // V tile: all waves stage + transpose into [hd][key].
        for (int i = 0; i < 8; ++i) {
            int f = tid + i * 256;              // 0..2047 uint4 slots
            int key = f >> 4;
            int d0  = (f & 15) << 3;
            size_t g = hbase + (size_t)(kb * 128 + key) * HS + d0;
            union { uint4 q; unsigned short s[8]; } u;
            u.q = *(const uint4*)(Vb + g);
            for (int e = 0; e < 8; ++e) Vt[(d0 + e) * KST + key] = u.s[e];
            if (kb + 1 < nkb) {                 // prefetch next key block
                __builtin_prefetch((const void*)(Vb + g + (size_t)128 * HS), 0, 1);
                __builtin_prefetch((const void*)(Kb + g + (size_t)128 * HS), 0, 1);
            }
        }
        __syncthreads();

        // S = Q K^T : 8 key tiles of 16, K-depth 128 (4 WMMA steps each).
        v8f sacc[8];
        for (int nt = 0; nt < 8; ++nt) {
            sacc[nt] = zero8();
            for (int ks = 0; ks < 4; ++ks) {
                const unsigned short* p = &Ksh[(nt * 16 + l16) * HS + ks * 32 + 8 * half];
                sacc[nt] = wmma_bf16(qf[ks], frag2(p, p + 16), sacc[nt]);
            }
        }

        // Online softmax (base-2 domain), per-row stats across each 16-lane half.
        unsigned short* pw = &Psh[wave * 16 * KST];
        for (int r = 0; r < 8; ++r) {
            float mx = sacc[0][r];
            for (int nt = 1; nt < 8; ++nt) mx = fmaxf(mx, sacc[nt][r]);
            for (int off = 1; off < 16; off <<= 1) mx = fmaxf(mx, __shfl_xor(mx, off, 32));
            float mn = fmaxf(mrow[r], mx * SL);
            float corr = __builtin_amdgcn_exp2f(mrow[r] - mn);
            mrow[r] = mn;
            lrow[r] *= corr;
            for (int ot = 0; ot < 8; ++ot) oacc[ot][r] *= corr;
            float ps = 0.f;
            int row = r + 8 * half;
            for (int nt = 0; nt < 8; ++nt) {
                float p = __builtin_amdgcn_exp2f(sacc[nt][r] * SL - mn);
                ps += p;
                pw[row * KST + nt * 16 + l16] = f2bf(p);  // C-layout -> LDS
            }
            for (int off = 1; off < 16; off <<= 1) ps += __shfl_xor(ps, off, 32);
            lrow[r] += ps;
        }
        // Per-wave cross-lane LDS RAW: drain ds stores before fragment re-read.
        asm volatile("s_wait_dscnt 0" ::: "memory");

        // O += P V : P re-read in A-matrix layout, V fragments from Vt.
        for (int ks = 0; ks < 4; ++ks) {
            const unsigned short* pp = &pw[l16 * KST + ks * 32 + 8 * half];
            v16bf pf = frag2(pp, pp + 16);
            for (int ot = 0; ot < 8; ++ot) {
                const unsigned short* vp = &Vt[(ot * 16 + l16) * KST + ks * 32 + 8 * half];
                oacc[ot] = wmma_bf16(pf, frag2(vp, vp + 16), oacc[ot]);
            }
        }
    }

    // Normalize and scatter to [B,S,H] f32.
    const int b = bh / nh, h = bh % nh;
    const int H = nh * HS;
    for (int r = 0; r < 8; ++r) {
        float inv = 1.0f / lrow[r];
        int s = qrow + r + 8 * half;
        for (int ot = 0; ot < 8; ++ot) {
            int d = ot * 16 + l16;
            Ob[(size_t)(b * S + s) * H + h * HS + d] = oacc[ot][r] * inv;
        }
    }
}

// ---------------------------------------------------------------------------
extern "C" void kernel_launch(void* const* d_in, const int* in_sizes, int n_in,
                              void* d_out, int out_size, void* d_ws, size_t ws_size,
                              hipStream_t stream) {
    const float* X  = (const float*)d_in[0];
    const float* Wq = (const float*)d_in[1];
    const float* Wk = (const float*)d_in[2];
    const float* Wv = (const float*)d_in[3];
    const float* Wo = (const float*)d_in[4];

    const int B = 2, S = 2048, H = 2048, NH = 16;
    const int M = B * S;  // 4096

    unsigned short* Qb = (unsigned short*)d_ws;
    unsigned short* Kb = Qb + (size_t)M * H;
    unsigned short* Vb = Kb + (size_t)M * H;
    float*          AO = (float*)(Vb + (size_t)M * H);

    dim3 gg(H / 128, M / 128);
    gemm_xwt_kernel<1><<<gg, 256, 0, stream>>>(X, Wq, Qb, M, H, H);
    gemm_xwt_kernel<1><<<gg, 256, 0, stream>>>(X, Wk, Kb, M, H, H);
    gemm_xwt_kernel<1><<<gg, 256, 0, stream>>>(X, Wv, Vb, M, H, H);
    attn_kernel<<<B * NH * (S / 128), 256, 0, stream>>>(Qb, Kb, Vb, AO, S, NH);
    gemm_xwt_kernel<0><<<gg, 256, 0, stream>>>(AO, Wo, (float*)d_out, M, H, H);
}
